// MMoE_51539607717
// MI455X (gfx1250) — compile-verified, matrix operand order checked by reading, not run on hardware
//
#include <hip/hip_runtime.h>
#include <stdint.h>

// ---------------------------------------------------------------------------
// MMoE on gfx1250: bf16 WMMA GEMM pipeline, async-to-LDS staging
//  - prep: x -> bf16, W -> transposed bf16 (one pass, off the critical loop)
//  - GEMM: 128x128 block, 8 waves (4Mx2N), wave tile 32x64 = 8 WMMAs/K-step
//  - staging via global_load_async_to_lds_b128 (ASYNCcnt), double-buffered LDS
// ---------------------------------------------------------------------------

#define B_ROWS 8192
#define D_IN   1024
#define N_EXP  8
#define N_TASK 4
#define H1     1024
#define H2     512
#define H3     256

typedef __attribute__((ext_vector_type(16))) __bf16 v16bf;
typedef __attribute__((ext_vector_type(8)))  float  v8f;
typedef __attribute__((ext_vector_type(4)))  unsigned int v4u;
typedef __attribute__((ext_vector_type(2)))  unsigned int v2u;

__device__ __forceinline__ unsigned short f2bf(float f) {
    unsigned int u = __float_as_uint(f);
    u += 0x7FFFu + ((u >> 16) & 1u);           // round to nearest even
    return (unsigned short)(u >> 16);
}
__device__ __forceinline__ float bf2f(unsigned short h) {
    return __uint_as_float(((unsigned int)h) << 16);
}
__device__ __forceinline__ unsigned int pack_bf2(float a, float b) {
    return (unsigned int)f2bf(a) | ((unsigned int)f2bf(b) << 16);
}

// async copy of 16 bytes: global -> LDS (per lane), tracked by ASYNCcnt
__device__ __forceinline__ void async_b128(unsigned lds_off, const void* gaddr) {
    asm volatile("global_load_async_to_lds_b128 %0, %1, off"
                 :: "v"(lds_off), "v"(gaddr) : "memory");
}
__device__ __forceinline__ void wait_async0() {
    asm volatile("s_wait_asynccnt 0x0" ::: "memory");
}

// ---------------------------------------------------------------------------
// Prep: fp32 -> bf16 (4 elements / thread)
// ---------------------------------------------------------------------------
__global__ void cvt_f32_bf16(const float* __restrict__ in,
                             unsigned short* __restrict__ out) {
    const int i = (blockIdx.x * blockDim.x + threadIdx.x) * 4;
    v2u p = { pack_bf2(in[i], in[i + 1]), pack_bf2(in[i + 2], in[i + 3]) };
    *(v2u*)(out + i) = p;
}

// ---------------------------------------------------------------------------
// Prep: W[e][K][N] fp32 -> Wt[e][N][K] bf16  (32x32 LDS tile transpose)
// ---------------------------------------------------------------------------
__global__ void transpose_cvt(const float* __restrict__ W,
                              unsigned short* __restrict__ Wt,
                              int K, int N) {
    __shared__ unsigned short t[32][33];
    const int n0 = blockIdx.x * 32;
    const int k0 = blockIdx.y * 32;
    const float*    Wp  = W  + (size_t)blockIdx.z * K * N;
    unsigned short* Wtp = Wt + (size_t)blockIdx.z * K * N;

    {
        const int kk = threadIdx.x >> 3;
        const int nn = (threadIdx.x & 7) * 4;
        const float* src = Wp + (size_t)(k0 + kk) * N + (n0 + nn);
#pragma unroll
        for (int j = 0; j < 4; ++j) t[kk][nn + j] = f2bf(src[j]);
    }
    __syncthreads();
    {
        const int nn = threadIdx.x >> 3;
        const int kk = (threadIdx.x & 7) * 4;
        unsigned short* dst = Wtp + (size_t)(n0 + nn) * K + (k0 + kk);
#pragma unroll
        for (int j = 0; j < 4; ++j) dst[j] = t[kk + j][nn];
    }
}

// ---------------------------------------------------------------------------
// Gate: logits = x @ gate_w  [B,32] -> softmax over groups of 8 (experts)
// ---------------------------------------------------------------------------
__global__ void gate_kernel(const float* __restrict__ x,
                            const float* __restrict__ gw,
                            float* __restrict__ gates) {
    const int wave = threadIdx.x >> 5;
    const int lane = threadIdx.x & 31;
    const int b    = blockIdx.x * 8 + wave;

    const float* xr = x + (size_t)b * D_IN;
    float acc = 0.f;
#pragma unroll 4
    for (int k = 0; k < D_IN; ++k)
        acc = fmaf(xr[k], gw[k * 32 + lane], acc);

    float mx = acc;
    mx = fmaxf(mx, __shfl_xor(mx, 1, 8));
    mx = fmaxf(mx, __shfl_xor(mx, 2, 8));
    mx = fmaxf(mx, __shfl_xor(mx, 4, 8));
    float p = __expf(acc - mx);
    float s = p;
    s += __shfl_xor(s, 1, 8);
    s += __shfl_xor(s, 2, 8);
    s += __shfl_xor(s, 4, 8);
    gates[(size_t)b * 32 + lane] = p / s;
}

// ---------------------------------------------------------------------------
// GEMM + bias + ReLU, bf16 in / bf16 out:
//   Out[M x N] = relu(A[M x K] * Wt[N x K]^T + bias)
// ---------------------------------------------------------------------------
#define BM 128
#define BN 128
#define BK 32
#define LDSK 40   // padded K stride (ushorts)

__global__ __launch_bounds__(256)
void gemm_bf16(const unsigned short* __restrict__ A,    // [M][K] bf16
               const unsigned short* __restrict__ Wt,   // [N][K] bf16
               const float*          __restrict__ bias, // [N] fp32
               unsigned short*       __restrict__ Out,  // [M][ldo] bf16
               int M, int K, int N, int ldo) {
    __shared__ unsigned short Asm[2][BM][LDSK];
    __shared__ unsigned short Bsm[2][BN][LDSK];

    const int tid  = threadIdx.x;
    const int lane = tid & 31;
    const int wave = tid >> 5;
    const int wm   = wave >> 1;        // 0..3 -> M offset wm*32
    const int wn   = wave & 1;         // 0..1 -> N offset wn*64
    const int m0   = blockIdx.y * BM;
    const int n0   = blockIdx.x * BN;

    const int lh = lane & 15;
    const int hi = lane >> 4;
    const int c0 = hi << 3;  // K chunk base: lane<16 -> {0-7,16-23}, lane>=16 -> {8-15,24-31}

    v8f c[2][4];
#pragma unroll
    for (int i = 0; i < 2; ++i)
#pragma unroll
        for (int j = 0; j < 4; ++j)
            c[i][j] = (v8f){0.f, 0.f, 0.f, 0.f, 0.f, 0.f, 0.f, 0.f};

    // staging: thread t handles row (t>>1) of both tiles, 16-elem half (t&1)
    const int row = tid >> 1;              // 0..127
    const int ks  = (tid & 1) * 16;        // 0 or 16

    const unsigned aoff0 = (unsigned)(uintptr_t)&Asm[0][row][ks];
    const unsigned aoff1 = (unsigned)(uintptr_t)&Asm[1][row][ks];
    const unsigned boff0 = (unsigned)(uintptr_t)&Bsm[0][row][ks];
    const unsigned boff1 = (unsigned)(uintptr_t)&Bsm[1][row][ks];

    const unsigned short* gA = A  + (size_t)(m0 + row) * K + ks;
    const unsigned short* gB = Wt + (size_t)(n0 + row) * K + ks;

    // prologue: stage tile 0 into buffer 0
    async_b128(aoff0,      gA);
    async_b128(aoff0 + 16, gA + 8);
    async_b128(boff0,      gB);
    async_b128(boff0 + 16, gB + 8);
    wait_async0();
    __syncthreads();

    const int nsteps = K / BK;
    for (int s = 0; s < nsteps; ++s) {
        const int buf = s & 1;

        // issue async staging for step s+1 into the other buffer
        if (s + 1 < nsteps) {
            const unsigned short* pA = gA + (size_t)(s + 1) * BK;
            const unsigned short* pB = gB + (size_t)(s + 1) * BK;
            const unsigned ao = buf ? aoff0 : aoff1;
            const unsigned bo = buf ? boff0 : boff1;
            async_b128(ao,      pA);
            async_b128(ao + 16, pA + 8);
            async_b128(bo,      pB);
            async_b128(bo + 16, pB + 8);
        }

        // fragments from current buffer
        v16bf afrag[2], bfrag[4];
#pragma unroll
        for (int tm = 0; tm < 2; ++tm) {
            const unsigned short* src = &Asm[buf][wm * 32 + tm * 16 + lh][c0];
            ((v4u*)&afrag[tm])[0] = *(const v4u*)(src);
            ((v4u*)&afrag[tm])[1] = *(const v4u*)(src + 16);
        }
#pragma unroll
        for (int tn = 0; tn < 4; ++tn) {
            const unsigned short* src = &Bsm[buf][wn * 64 + tn * 16 + lh][c0];
            ((v4u*)&bfrag[tn])[0] = *(const v4u*)(src);
            ((v4u*)&bfrag[tn])[1] = *(const v4u*)(src + 16);
        }

        // 8 WMMAs (A fragment reused across the tn sweep)
#pragma unroll
        for (int tm = 0; tm < 2; ++tm)
#pragma unroll
            for (int tn = 0; tn < 4; ++tn)
                c[tm][tn] = __builtin_amdgcn_wmma_f32_16x16x32_bf16(
                    false, afrag[tm], false, bfrag[tn],
                    (short)0, c[tm][tn], false, false);

        wait_async0();     // own async stores to LDS complete
        __syncthreads();   // all waves' staging committed
    }

    // epilogue: bias + relu + bf16 store
#pragma unroll
    for (int tm = 0; tm < 2; ++tm) {
#pragma unroll
        for (int tn = 0; tn < 4; ++tn) {
            const int gn = n0 + wn * 64 + tn * 16 + lh;
            const float bn = bias[gn];
#pragma unroll
            for (int r = 0; r < 8; ++r) {
                const int gm = m0 + wm * 32 + tm * 16 + hi * 8 + r;
                float v = c[tm][tn][r] + bn;
                v = fmaxf(v, 0.f);
                Out[(size_t)gm * ldo + gn] = f2bf(v);
            }
        }
    }
}

// ---------------------------------------------------------------------------
// Mix: out[t][b][d] = sum_e h3[b][e][d] * gates[b][t*8+e]
// ---------------------------------------------------------------------------
__global__ void mix_kernel(const unsigned short* __restrict__ h3,
                           const float* __restrict__ gates,
                           float* __restrict__ out) {
    const int idx = blockIdx.x * blockDim.x + threadIdx.x;  // over B*256
    const int b = idx >> 8;
    const int d = idx & 255;
    const unsigned short* hp = h3 + ((size_t)b << 11) + d;  // b*E*H3 + d
    const float* g = gates + (size_t)b * 32;

    float acc0 = 0.f, acc1 = 0.f, acc2 = 0.f, acc3 = 0.f;
#pragma unroll
    for (int e = 0; e < 8; ++e) {
        const float hv = bf2f(hp[e << 8]);
        acc0 = fmaf(hv, g[e],      acc0);
        acc1 = fmaf(hv, g[8 + e],  acc1);
        acc2 = fmaf(hv, g[16 + e], acc2);
        acc3 = fmaf(hv, g[24 + e], acc3);
    }
    const size_t stride = (size_t)B_ROWS * H3;
    out[0 * stride + idx] = acc0;
    out[1 * stride + idx] = acc1;
    out[2 * stride + idx] = acc2;
    out[3 * stride + idx] = acc3;
}

// ---------------------------------------------------------------------------
// Launch
// ---------------------------------------------------------------------------
extern "C" void kernel_launch(void* const* d_in, const int* in_sizes, int n_in,
                              void* d_out, int out_size, void* d_ws, size_t ws_size,
                              hipStream_t stream) {
    const float* x  = (const float*)d_in[0];
    const float* W1 = (const float*)d_in[1];
    const float* b1 = (const float*)d_in[2];
    const float* W2 = (const float*)d_in[3];
    const float* b2 = (const float*)d_in[4];
    const float* W3 = (const float*)d_in[5];
    const float* b3 = (const float*)d_in[6];
    const float* gw = (const float*)d_in[7];
    float* out = (float*)d_out;

    // workspace layout (MB offsets)
    char* ws = (char*)d_ws;
    float*          gates = (float*)ws;                                  //  0: 1 MB
    unsigned short* xb    = (unsigned short*)(ws + (size_t)(1  << 20));  //  1: 16 MB [B,DIN] bf16
    unsigned short* W1t   = (unsigned short*)(ws + (size_t)(17 << 20));  // 17: 16 MB [E,H1,DIN] bf16
    unsigned short* W2t   = (unsigned short*)(ws + (size_t)(33 << 20));  // 33:  8 MB [E,H2,H1] bf16
    unsigned short* W3t   = (unsigned short*)(ws + (size_t)(41 << 20));  // 41:  2 MB [E,H3,H2] bf16
    unsigned short* h1    = (unsigned short*)(ws + (size_t)(43 << 20));  // 43: 16 MB [B,H1] bf16
    unsigned short* h2    = (unsigned short*)(ws + (size_t)(59 << 20));  // 59:  8 MB [B,H2] bf16
    unsigned short* h3    = (unsigned short*)(ws + (size_t)(67 << 20));  // 67: 32 MB [B,E,H3] bf16

    // prep passes (off the critical GEMM loop)
    cvt_f32_bf16<<<(B_ROWS * D_IN) / 1024, 256, 0, stream>>>(x, xb);
    transpose_cvt<<<dim3(H1 / 32, D_IN / 32, N_EXP), 256, 0, stream>>>(W1, W1t, D_IN, H1);
    transpose_cvt<<<dim3(H2 / 32, H1 / 32,  N_EXP), 256, 0, stream>>>(W2, W2t, H1, H2);
    transpose_cvt<<<dim3(H3 / 32, H2 / 32,  N_EXP), 256, 0, stream>>>(W3, W3t, H2, H3);

    gate_kernel<<<B_ROWS / 8, 256, 0, stream>>>(x, gw, gates);

    for (int e = 0; e < N_EXP; ++e) {
        gemm_bf16<<<dim3(H1 / BN, B_ROWS / BM), 256, 0, stream>>>(
            xb, W1t + (size_t)e * D_IN * H1, b1 + (size_t)e * H1,
            h1, B_ROWS, D_IN, H1, H1);
        gemm_bf16<<<dim3(H2 / BN, B_ROWS / BM), 256, 0, stream>>>(
            h1, W2t + (size_t)e * H1 * H2, b2 + (size_t)e * H2,
            h2, B_ROWS, H1, H2, H2);
        gemm_bf16<<<dim3(H3 / BN, B_ROWS / BM), 256, 0, stream>>>(
            h2, W3t + (size_t)e * H2 * H3, b3 + (size_t)e * H3,
            h3 + (size_t)e * H3, B_ROWS, H2, H3, N_EXP * H3);
    }

    mix_kernel<<<(B_ROWS * H3) / 256, 256, 0, stream>>>(h3, gates, out);
}
